// LSTM_DSSM_63290638074440
// MI455X (gfx1250) — matrix operand order repaired and checked by reading, not compiled, not run on hardware
//
#include <hip/hip_runtime.h>
#include <hip/hip_bf16.h>

// ---------------- problem constants ----------------
#define VOCABN   50000
#define EMBD     128      // E
#define HID      256      // H
#define GATE     1024     // 4*H, order: i | f | c | o
#define SLEN     64       // timesteps
#define NSEQ     4096     // 128 batch * 32 (1 query + 31 docs)
#define BATCHN   128
#define NDOC     31
#define MT       16       // sequences per workgroup

typedef __attribute__((ext_vector_type(16))) _Float16 v16h;
typedef __attribute__((ext_vector_type(8)))  float    v8f;

union FragH { v16h v; uint4 q[2]; };

// ---------------- prep: f32->f16 converts, weight transpose, token windows ----
__global__ void prep_kernel(const float* __restrict__ emb,
                            const float* __restrict__ k_f,  const float* __restrict__ rk_f,
                            const float* __restrict__ k_b,  const float* __restrict__ rk_b,
                            const int*   __restrict__ inputs,
                            _Float16* __restrict__ emb16,
                            _Float16* __restrict__ kT16,    // [2][GATE][EMBD]
                            _Float16* __restrict__ rkT16,   // [2][GATE][HID]
                            int* __restrict__ tokens)       // [NSEQ][SLEN]
{
    const int idx0   = blockIdx.x * blockDim.x + threadIdx.x;
    const int stride = gridDim.x * blockDim.x;

    for (int i = idx0; i < VOCABN * EMBD; i += stride)
        emb16[i] = (_Float16)emb[i];

    for (int i = idx0; i < GATE * EMBD; i += stride) {
        int g = i >> 7, e = i & (EMBD - 1);
        kT16[i]               = (_Float16)k_f[e * GATE + g];
        kT16[GATE * EMBD + i] = (_Float16)k_b[e * GATE + g];
    }

    for (int i = idx0; i < GATE * HID; i += stride) {
        int g = i >> 8, h = i & (HID - 1);
        rkT16[i]              = (_Float16)rk_f[h * GATE + g];
        rkT16[GATE * HID + i] = (_Float16)rk_b[h * GATE + g];
    }

    for (int i = idx0; i < NSEQ * SLEN; i += stride) {
        int n = i >> 6, t = i & (SLEN - 1);
        int b = n >> 5, j = n & 31;                 // j==0: query, else doc window j-1
        int tok = (j == 0) ? inputs[b * 158 + t]
                           : inputs[b * 158 + 64 + (j - 1) + t];
        tokens[i] = tok;
    }
}

// ---------------- fused bidirectional peephole-LSTM kernel ----------------
// grid = (NSEQ/MT, 2 directions), 256 threads = 8 waves.
// Wave w owns hidden slice [w*32, w*32+32) across all 4 gates -> 8 C-tiles.
__global__ __launch_bounds__(256)
void lstm_kernel(const int*      __restrict__ tokens,
                 const _Float16* __restrict__ emb16,
                 const _Float16* __restrict__ kT16,
                 const _Float16* __restrict__ rkT16,
                 const float* __restrict__ b_f,   const float* __restrict__ b_b,
                 const float* __restrict__ wci_f, const float* __restrict__ wcf_f,
                 const float* __restrict__ wco_f,
                 const float* __restrict__ wci_b, const float* __restrict__ wcf_b,
                 const float* __restrict__ wco_b,
                 float* __restrict__ hfinal)      // [2][NSEQ][HID]
{
    __shared__ _Float16 emb_s[MT][EMBD + 8];   // 272 B rows (16B aligned)
    __shared__ _Float16 h_s[MT][HID + 8];      // 528 B rows (16B aligned)
    __shared__ int      tok_s[MT];

    const int dir  = blockIdx.y;
    const int seq0 = blockIdx.x * MT;
    const int tid  = threadIdx.x;
    const int wave = tid >> 5;
    const int lane = tid & 31;
    const int ln16 = lane & 15;
    const int hi   = lane >> 4;           // lane half
    const int kbA  = hi ? 8 : 0;          // A-frag K base per ISA layout
    const int kbB  = hi ? 8 : 0;          // B-frag K base per ISA layout
    const int mbase = hi * 8;             // C-tile rows owned by this lane

    const float* bias = dir ? b_b  : b_f;
    const float* wci  = dir ? wci_b : wci_f;
    const float* wcf  = dir ? wcf_b : wcf_f;
    const float* wco  = dir ? wco_b : wco_f;
    const _Float16* kT  = kT16  + (size_t)dir * GATE * EMBD;
    const _Float16* rkT = rkT16 + (size_t)dir * GATE * HID;

    // Per-lane tile columns + bias. Tile tt: gate g=tt>>1, half=tt&1.
    int   gcol[8];
    float biasv[8];
#pragma unroll
    for (int tt = 0; tt < 8; ++tt) {
        int col   = (tt >> 1) * HID + wave * 32 + (tt & 1) * 16 + ln16;
        gcol[tt]  = col;
        biasv[tt] = bias[col];
    }
    // Peephole weights for this lane's hidden unit (one per half).
    float pwi[2], pwf[2], pwo[2];
#pragma unroll
    for (int half = 0; half < 2; ++half) {
        int hh = wave * 32 + half * 16 + ln16;
        pwi[half] = wci[hh]; pwf[half] = wcf[hh]; pwo[half] = wco[hh];
    }

    // Registered LSTM state: lane owns (m=mbase+r, hh(half)) for r=0..7.
    float cst[2][8], hst[2][8];
#pragma unroll
    for (int half = 0; half < 2; ++half)
#pragma unroll
        for (int r = 0; r < 8; ++r) { cst[half][r] = 0.f; hst[half][r] = 0.f; }

    for (int i = tid; i < MT * (HID + 8); i += 256)
        ((_Float16*)h_s)[i] = (_Float16)0.f;

    for (int t = 0; t < SLEN; ++t) {
        const int tt_idx = dir ? (SLEN - 1 - t) : t;

        // Launder a zero OFFSET (not the pointer!) each iteration: the base
        // stays traceable to the kernarg so loads remain global_load_b128
        // (addrspace(1)), but the addresses become loop-variant so LICM cannot
        // hoist the loop-invariant weight fragments and spill them to scratch.
        size_t lz = 0;
        asm volatile("" : "+s"(lz));
        const _Float16* kTc  = kT  + lz;
        const _Float16* rkTc = rkT + lz;

        __syncthreads();   // prev-step h_s writes visible; emb_s free to reuse

        // Gather this step's embedding tile (16 rows x 128 f16) + tokens.
        {
            const int row = tid >> 4;          // 0..15 sequence row
            const int ch  = tid & 15;          // 16B chunk within row
            const int tok = tokens[(seq0 + row) * SLEN + tt_idx];
            if (ch == 0) tok_s[row] = tok;
            const uint4* src = (const uint4*)(emb16 + (size_t)tok * EMBD) + ch;
            *(uint4*)&emb_s[row][ch * 8] = *src;
            if (t + 1 < SLEN) {                // prefetch next step's rows (global_prefetch_b8)
                const int tn   = dir ? (SLEN - 2 - t) : (t + 1);
                const int tok2 = tokens[(seq0 + row) * SLEN + tn];
                __builtin_prefetch(emb16 + (size_t)tok2 * EMBD + ch * 8, 0, 1);
            }
        }
        __syncthreads();

        // z = bias + emb@k + h@rk, 8 output tiles per wave.
        v8f acc[8];
#pragma unroll
        for (int tt = 0; tt < 8; ++tt) {
            v8f z;
#pragma unroll
            for (int r = 0; r < 8; ++r) z[r] = biasv[tt];
            acc[tt] = z;
        }

        // Input projection: K = EMBD = 128 -> 4 chunks of 32.
        // Batch all 8 B-fragment loads (16x global_load_b128 clause) per
        // k-chunk so L2 latency hides behind the 8 WMMAs.
#pragma unroll
        for (int kc = 0; kc < EMBD / 32; ++kc) {
            FragH a;
            a.q[0] = *(const uint4*)&emb_s[ln16][kc * 32 + kbA];
            a.q[1] = *(const uint4*)&emb_s[ln16][kc * 32 + kbA + 16];
            FragH b[8];
#pragma unroll
            for (int tt = 0; tt < 8; ++tt) {
                const _Float16* bp = kTc + (size_t)gcol[tt] * EMBD + kc * 32 + kbB;
                b[tt].q[0] = *(const uint4*)bp;
                b[tt].q[1] = *(const uint4*)(bp + 16);
            }
#pragma unroll
            for (int tt = 0; tt < 8; ++tt)
                acc[tt] = __builtin_amdgcn_wmma_f32_16x16x32_f16(
                    false, a.v, false, b[tt].v, (short)0, acc[tt], false, false);
        }
        // Recurrence: K = HID = 256 -> 8 chunks of 32.
#pragma unroll
        for (int kc = 0; kc < HID / 32; ++kc) {
            FragH a;
            a.q[0] = *(const uint4*)&h_s[ln16][kc * 32 + kbA];
            a.q[1] = *(const uint4*)&h_s[ln16][kc * 32 + kbA + 16];
            FragH b[8];
#pragma unroll
            for (int tt = 0; tt < 8; ++tt) {
                const _Float16* bp = rkTc + (size_t)gcol[tt] * HID + kc * 32 + kbB;
                b[tt].q[0] = *(const uint4*)bp;
                b[tt].q[1] = *(const uint4*)(bp + 16);
            }
#pragma unroll
            for (int tt = 0; tt < 8; ++tt)
                acc[tt] = __builtin_amdgcn_wmma_f32_16x16x32_f16(
                    false, a.v, false, b[tt].v, (short)0, acc[tt], false, false);
        }

        __syncthreads();   // all waves done reading h_s before overwrite

        // Gate math: all four gates for (m,hh) are resident in THIS lane.
#pragma unroll
        for (int half = 0; half < 2; ++half) {
            const int hh = wave * 32 + half * 16 + ln16;
#pragma unroll
            for (int r = 0; r < 8; ++r) {
                const int m  = mbase + r;
                float zi = acc[0 * 2 + half][r];
                float zf = acc[1 * 2 + half][r];
                float zc = acc[2 * 2 + half][r];
                float zo = acc[3 * 2 + half][r];
                float c  = cst[half][r];
                float ig = 1.f / (1.f + __expf(-(zi + c * pwi[half])));
                float fg = 1.f / (1.f + __expf(-(zf + c * pwf[half])));
                float cn = fg * c + ig * tanhf(zc);
                float og = 1.f / (1.f + __expf(-(zo + cn * pwo[half])));
                float hn = og * tanhf(cn);
                bool  msk = (tok_s[m] != 0);
                float hv = msk ? hn : hst[half][r];
                float cv = msk ? cn : c;
                hst[half][r] = hv;
                cst[half][r] = cv;
                h_s[m][hh] = (_Float16)hv;
            }
        }
    }

    // Write final hidden state.
#pragma unroll
    for (int half = 0; half < 2; ++half) {
        const int hh = wave * 32 + half * 16 + ln16;
#pragma unroll
        for (int r = 0; r < 8; ++r) {
            const int m = mbase + r;
            hfinal[((size_t)dir * NSEQ + seq0 + m) * HID + hh] = hst[half][r];
        }
    }
}

// ---------------- cosine similarity + softmax ----------------
__global__ void scores_kernel(const float* __restrict__ hfinal,
                              float* __restrict__ out)
{
    __shared__ float sh[32][HID];   // averaged hidden states for this batch
    __shared__ float snn[32], sdp[32], sc[32], ssum;

    const int b   = blockIdx.x;
    const int tid = threadIdx.x;
    const float* hf = hfinal;
    const float* hb = hfinal + (size_t)NSEQ * HID;

    for (int i = tid; i < 32 * HID; i += 256) {
        int j = i >> 8, k = i & (HID - 1);
        size_t seq = (size_t)b * 32 + j;
        sh[j][k] = 0.5f * (hf[seq * HID + k] + hb[seq * HID + k]);
    }
    __syncthreads();

    if (tid < 32) {
        float nn = 0.f, dp = 0.f;
        for (int k = 0; k < HID; ++k) {
            float v = sh[tid][k];
            nn += v * v;
            dp += v * sh[0][k];
        }
        snn[tid] = nn;
        sdp[tid] = dp;
    }
    __syncthreads();

    if (tid >= 1 && tid < 32) {
        float denom = sqrtf(snn[0]) * sqrtf(snn[tid]);
        sc[tid] = fabsf(sdp[tid]) / denom;
    }
    __syncthreads();

    if (tid == 0) {
        float mx = -1e30f;
        for (int j = 1; j < 32; ++j) mx = fmaxf(mx, sc[j]);
        float s = 0.f;
        for (int j = 1; j < 32; ++j) { float e = __expf(sc[j] - mx); sc[j] = e; s += e; }
        ssum = s;
    }
    __syncthreads();

    if (tid < NDOC) out[b * NDOC + tid] = sc[tid + 1] / ssum;
}

// ---------------- launcher ----------------
extern "C" void kernel_launch(void* const* d_in, const int* in_sizes, int n_in,
                              void* d_out, int out_size, void* d_ws, size_t ws_size,
                              hipStream_t stream) {
    (void)in_sizes; (void)n_in; (void)out_size; (void)ws_size;

    const int*   inputs = (const int*)  d_in[0];
    const float* emb    = (const float*)d_in[1];
    const float* k_f    = (const float*)d_in[2];
    const float* rk_f   = (const float*)d_in[3];
    const float* b_f    = (const float*)d_in[4];
    const float* wci_f  = (const float*)d_in[5];
    const float* wcf_f  = (const float*)d_in[6];
    const float* wco_f  = (const float*)d_in[7];
    const float* k_b    = (const float*)d_in[8];
    const float* rk_b   = (const float*)d_in[9];
    const float* b_b    = (const float*)d_in[10];
    const float* wci_b  = (const float*)d_in[11];
    const float* wcf_b  = (const float*)d_in[12];
    const float* wco_b  = (const float*)d_in[13];

    char* w = (char*)d_ws;
    _Float16* emb16 = (_Float16*)w;  w += (size_t)VOCABN * EMBD * 2;  // 12.8 MB
    _Float16* kT16  = (_Float16*)w;  w += (size_t)2 * GATE * EMBD * 2;
    _Float16* rkT16 = (_Float16*)w;  w += (size_t)2 * GATE * HID * 2;
    int*      toks  = (int*)w;       w += (size_t)NSEQ * SLEN * 4;
    float*    hfin  = (float*)w;     w += (size_t)2 * NSEQ * HID * 4;

    prep_kernel<<<512, 256, 0, stream>>>(emb, k_f, rk_f, k_b, rk_b, inputs,
                                         emb16, kT16, rkT16, toks);

    lstm_kernel<<<dim3(NSEQ / MT, 2), 256, 0, stream>>>(
        toks, emb16, kT16, rkT16, b_f, b_b,
        wci_f, wcf_f, wco_f, wci_b, wcf_b, wco_b, hfin);

    scores_kernel<<<BATCHN, 256, 0, stream>>>(hfin, (float*)d_out);
}
